// UnifiedLIV_84911503442800
// MI455X (gfx1250) — matrix operand order repaired and checked by reading, not compiled, not run on hardware
//
#include <hip/hip_runtime.h>
#include <math.h>

// Problem constants (from reference)
#define DIMD     512
#define INTERNAL 8192
#define SEQ      2048
#define BATCH    2

// Kernel 1 tiling: one WG = (batch b, 64-channel tile), persistent over all of S.
#define NTILE    64      // channels per WG
#define SCHUNK   32      // sequence rows per GEMM chunk
#define THREADS1 128     // 4 waves (wave32)
#define XSTR     (DIMD + 8)  // padded LDS row stride (bf16 elems) -> conflict-free frags

// Kernel 2 tiling (output projection GEMM)
#define MT2      32
#define NT2      64
#define KB2      64
#define THREADS2 128
#define K2STR    (KB2 + 8)

typedef __attribute__((ext_vector_type(16))) __bf16 v16bf;
typedef __attribute__((ext_vector_type(8)))  __bf16 v8bf;
typedef __attribute__((ext_vector_type(8)))  float  v8f;

union V8F { v8f v; float f[8]; };

// Assemble a 16-element bf16 fragment from two 16B LDS loads.
__device__ __forceinline__ v16bf load_frag16(const __bf16* p0, const __bf16* p1) {
    v8bf a0 = *(const v8bf*)p0;
    v8bf a1 = *(const v8bf*)p1;
    return __builtin_shufflevector(a0, a1, 0,1,2,3,4,5,6,7,8,9,10,11,12,13,14,15);
}

__device__ __forceinline__ float softplus_f(float v) {
    return (v > 20.0f) ? v : log1pf(__expf(v));
}

// ---------------------------------------------------------------------------
// Kernel 1: fused 4x projection GEMM (bf16 WMMA) + softplus/exp/discretize +
// sequential scan h_t = A_t*h_{t-1} + u_t, writing y = C*h (bf16) to workspace.
// Grid: (INTERNAL/NTILE, BATCH). Each WG keeps its 4 weight tiles resident in
// LDS (bf16) for the whole sequence; scan state h lives in registers.
// 316 KB LDS -> 1 WG/WGP (1 wave/SIMD), so the next chunk's x rows are
// software-prefetched (global_prefetch_b8) while the WMMA loop runs.
// ---------------------------------------------------------------------------
__global__ __launch_bounds__(THREADS1)
void ssm_featurize_scan_kernel(const float* __restrict__ x,
                               const float* __restrict__ Wdt, const float* __restrict__ bdt,
                               const float* __restrict__ WB,  const float* __restrict__ bB,
                               const float* __restrict__ WC,  const float* __restrict__ bC,
                               const float* __restrict__ WV,  const float* __restrict__ bV,
                               const float* __restrict__ Alog,
                               __bf16* __restrict__ y)
{
    extern __shared__ char smem[];
    __bf16* wl = (__bf16*)smem;                       // [4][NTILE][XSTR]  bf16 weight tiles
    __bf16* xl = wl + 4 * NTILE * XSTR;               // [SCHUNK][XSTR]    bf16 x chunk
    float*  sA = (float*)(xl + SCHUNK * XSTR);        // [SCHUNK][NTILE]   A_bar
    float*  sU = sA + SCHUNK * NTILE;                 // [SCHUNK][NTILE]   u = B_bar*V
    float*  sC = sU + SCHUNK * NTILE;                 // [SCHUNK][NTILE]   C

    const int tid  = threadIdx.x;
    const int wave = tid >> 5;
    const int lane = tid & 31;
    const int hi   = lane >> 4;     // lane half (WMMA layout)
    const int lq   = lane & 15;
    const int nb   = blockIdx.x * NTILE;  // channel tile base
    const int b    = blockIdx.y;          // batch

    // ---- Stage weight tiles once (fp32 -> bf16) ----
    const float* Wm[4] = {Wdt, WB, WC, WV};
#pragma unroll
    for (int m = 0; m < 4; ++m) {
        const float* src = Wm[m] + (size_t)nb * DIMD;
        __bf16* dst = wl + m * NTILE * XSTR;
        for (int idx = tid; idx < NTILE * (DIMD / 4); idx += THREADS1) {
            int r = idx / (DIMD / 4), c = (idx % (DIMD / 4)) * 4;
            const float4 v = *(const float4*)(src + (size_t)r * DIMD + c);
            __bf16* d = dst + r * XSTR + c;
            d[0] = (__bf16)v.x; d[1] = (__bf16)v.y; d[2] = (__bf16)v.z; d[3] = (__bf16)v.w;
        }
    }

    // Per-lane channel constants (lane column lq of this wave's 16-ch strip)
    const int nch = nb + wave * 16 + lq;
    const float bdt_l = bdt[nch], bB_l = bB[nch], bC_l = bC[nch], bV_l = bV[nch];
    const float Al_l  = Alog[nch];

    float hcarry = 0.0f;   // scan state (threads 0..63 own channel nb+tid)

    __syncthreads();

    for (int s0 = 0; s0 < SEQ; s0 += SCHUNK) {
        // ---- Stage x chunk (fp32 -> bf16) ----
        const float* xs = x + ((size_t)b * SEQ + s0) * DIMD;
        for (int idx = tid; idx < SCHUNK * (DIMD / 4); idx += THREADS1) {
            int r = idx / (DIMD / 4), c = (idx % (DIMD / 4)) * 4;
            const float4 v = *(const float4*)(xs + (size_t)r * DIMD + c);
            __bf16* d = xl + r * XSTR + c;
            d[0] = (__bf16)v.x; d[1] = (__bf16)v.y; d[2] = (__bf16)v.z; d[3] = (__bf16)v.w;
        }
        // Prefetch next chunk's x rows toward this WGP while WMMAs run.
        // 64KB = 1024 cachelines(64B); 128 threads x 8 lines each.
        if (s0 + SCHUNK < SEQ) {
            const char* nx = (const char*)(x + ((size_t)b * SEQ + s0 + SCHUNK) * DIMD);
#pragma unroll
            for (int p = 0; p < 8; ++p)
                __builtin_prefetch(nx + ((size_t)tid * 8 + p) * 64, 0, 3);
        }
        __syncthreads();

        // ---- 4 projections via WMMA: [32 x 512] x [512 x 64] per WG ----
        const v8f vzero = {0.f,0.f,0.f,0.f,0.f,0.f,0.f,0.f};
        V8F acc[4][2];
#pragma unroll
        for (int m = 0; m < 4; ++m) { acc[m][0].v = vzero; acc[m][1].v = vzero; }

        const int brow = (wave * 16 + lq) * XSTR;   // B operand: lane = channel
#pragma unroll 4
        for (int kk = 0; kk < DIMD; kk += 32) {
            // A fragments (M = s rows): lane=M, K 0-7/16-23 (hi=0) or 8-15/24-31 (hi=1)
            const __bf16* xr0 = xl + lq * XSTR + kk + hi * 8;
            const __bf16* xr1 = xl + (16 + lq) * XSTR + kk + hi * 8;
            v16bf a0 = load_frag16(xr0, xr0 + 16);
            v16bf a1 = load_frag16(xr1, xr1 + 16);
#pragma unroll
            for (int m = 0; m < 4; ++m) {
                // B fragment: K 0-7/16-23 (hi=0) or 8-15/24-31 (hi=1) of this lane's channel row
                const __bf16* wr = wl + m * NTILE * XSTR + brow + kk + hi * 8;
                v16bf bf = load_frag16(wr, wr + 16);
                acc[m][0].v = __builtin_amdgcn_wmma_f32_16x16x32_bf16(
                    false, a0, false, bf, (short)0, acc[m][0].v, false, false);
                acc[m][1].v = __builtin_amdgcn_wmma_f32_16x16x32_bf16(
                    false, a1, false, bf, (short)0, acc[m][1].v, false, false);
            }
        }

        // ---- Elementwise epilogue -> scan inputs in LDS ----
        // C/D layout: lane column n = lq (+16*wave), VGPR r = row r + 8*hi (+16*mt)
        const int ccol = wave * 16 + lq;
#pragma unroll
        for (int mt = 0; mt < 2; ++mt) {
#pragma unroll
            for (int r = 0; r < 8; ++r) {
                const int mrow = mt * 16 + hi * 8 + r;
                const float dt = softplus_f(acc[0][mt].f[r] + bdt_l);
                const float Ab = __expf(dt * Al_l);
                const float Bb = dt * (acc[1][mt].f[r] + bB_l);
                const float Cc = acc[2][mt].f[r] + bC_l;
                const float Vv = acc[3][mt].f[r] + bV_l;
                sA[mrow * NTILE + ccol] = Ab;
                sU[mrow * NTILE + ccol] = Bb * Vv;
                sC[mrow * NTILE + ccol] = Cc;
            }
        }
        __syncthreads();

        // ---- Sequential scan over the 32 chunk rows (1 thread per channel) ----
        if (tid < NTILE) {
            float h = hcarry;
            __bf16* yo = y + ((size_t)b * SEQ + s0) * INTERNAL + nb + tid;
#pragma unroll
            for (int t = 0; t < SCHUNK; ++t) {
                h = sA[t * NTILE + tid] * h + sU[t * NTILE + tid];
                yo[(size_t)t * INTERNAL] = (__bf16)(sC[t * NTILE + tid] * h);
            }
            hcarry = h;
        }
        __syncthreads();
    }
}

// ---------------------------------------------------------------------------
// Kernel 2: output projection out[m,d] = sum_n y[m,n]*W_out[d,n] + b_out[d]
// M=4096, N=512, K=8192.  LDS-tiled bf16 WMMA GEMM, fp32 accumulate/output.
// Grid: (M/32, 512/64), 128 threads (4 waves, each owns a 16-wide d strip).
// ---------------------------------------------------------------------------
__global__ __launch_bounds__(THREADS2)
void out_proj_kernel(const __bf16* __restrict__ y,
                     const float*  __restrict__ Wout,
                     const float*  __restrict__ bout,
                     float* __restrict__ out)
{
    __shared__ __attribute__((aligned(16))) __bf16 yl[MT2][K2STR];
    __shared__ __attribute__((aligned(16))) __bf16 wlo[NT2][K2STR];

    const int tid  = threadIdx.x;
    const int wave = tid >> 5;
    const int lane = tid & 31;
    const int hi   = lane >> 4;
    const int lq   = lane & 15;
    const int m0   = blockIdx.x * MT2;   // row base in flattened [B*S]
    const int d0   = blockIdx.y * NT2;   // output-dim base

    const v8f vzero = {0.f,0.f,0.f,0.f,0.f,0.f,0.f,0.f};
    V8F acc[2];
    acc[0].v = vzero; acc[1].v = vzero;

    for (int k0 = 0; k0 < INTERNAL; k0 += KB2) {
        // Stage y tile [32 x 64] (already bf16)
        for (int idx = tid; idx < MT2 * (KB2 / 8); idx += THREADS2) {
            int r = idx / (KB2 / 8), c = (idx % (KB2 / 8)) * 8;
            *(v8bf*)&yl[r][c] = *(const v8bf*)(y + (size_t)(m0 + r) * INTERNAL + k0 + c);
        }
        // Stage W_out tile [64 x 64] fp32 -> bf16 (K-contiguous rows: B-operand order)
        for (int idx = tid; idx < NT2 * (KB2 / 4); idx += THREADS2) {
            int r = idx / (KB2 / 4), c = (idx % (KB2 / 4)) * 4;
            const float4 v = *(const float4*)(Wout + (size_t)(d0 + r) * INTERNAL + k0 + c);
            __bf16* d = &wlo[r][c];
            d[0] = (__bf16)v.x; d[1] = (__bf16)v.y; d[2] = (__bf16)v.z; d[3] = (__bf16)v.w;
        }
        // Prefetch next K-panel (y rows and W_out rows) while this panel computes.
        if (k0 + KB2 < INTERNAL) {
            // y: 32 rows x 128B  -> 64 lines; W_out: 64 rows x 256B -> 256 lines
            const int r1 = tid & 31, r2 = tid >> 5;      // 32 rows x (1 line each of 128B)
            __builtin_prefetch((const char*)(y + (size_t)(m0 + r1) * INTERNAL + k0 + KB2) + r2 * 32, 0, 3);
            const int wr_ = tid >> 1, wc_ = (tid & 1) * 32;  // 64 rows x 2 halves (128B each)
            __builtin_prefetch((const char*)(Wout + (size_t)(d0 + wr_) * INTERNAL + k0 + KB2 + wc_), 0, 3);
        }
        __syncthreads();

#pragma unroll
        for (int kk = 0; kk < KB2; kk += 32) {
            const __bf16* xr0 = &yl[lq][kk + hi * 8];
            const __bf16* xr1 = &yl[16 + lq][kk + hi * 8];
            v16bf a0 = load_frag16(xr0, xr0 + 16);
            v16bf a1 = load_frag16(xr1, xr1 + 16);
            const __bf16* wr = &wlo[wave * 16 + lq][kk + hi * 8];
            v16bf bf = load_frag16(wr, wr + 16);
            acc[0].v = __builtin_amdgcn_wmma_f32_16x16x32_bf16(
                false, a0, false, bf, (short)0, acc[0].v, false, false);
            acc[1].v = __builtin_amdgcn_wmma_f32_16x16x32_bf16(
                false, a1, false, bf, (short)0, acc[1].v, false, false);
        }
        __syncthreads();
    }

    const int d  = d0 + wave * 16 + lq;
    const float bo = bout[d];
#pragma unroll
    for (int mt = 0; mt < 2; ++mt) {
#pragma unroll
        for (int r = 0; r < 8; ++r) {
            const int m = m0 + mt * 16 + hi * 8 + r;
            out[(size_t)m * DIMD + d] = acc[mt].f[r] + bo;
        }
    }
}

// ---------------------------------------------------------------------------
extern "C" void kernel_launch(void* const* d_in, const int* in_sizes, int n_in,
                              void* d_out, int out_size, void* d_ws, size_t ws_size,
                              hipStream_t stream) {
    const float* x    = (const float*)d_in[0];
    const float* Wdt  = (const float*)d_in[1];
    const float* bdt  = (const float*)d_in[2];
    const float* WB   = (const float*)d_in[3];
    const float* bB   = (const float*)d_in[4];
    const float* WC   = (const float*)d_in[5];
    const float* bC   = (const float*)d_in[6];
    const float* WV   = (const float*)d_in[7];
    const float* bV   = (const float*)d_in[8];
    const float* Alog = (const float*)d_in[9];
    const float* Wout = (const float*)d_in[10];
    const float* bout = (const float*)d_in[11];
    float* out   = (float*)d_out;
    __bf16* ybuf = (__bf16*)d_ws;   // [BATCH][SEQ][INTERNAL] bf16 = 64 MB

    // Kernel 1: fused projections + scan
    const size_t smem1 = (size_t)(4 * NTILE * XSTR + SCHUNK * XSTR) * sizeof(__bf16)
                       + (size_t)3 * SCHUNK * NTILE * sizeof(float);   // ~316 KB < 320 KB
    dim3 g1(INTERNAL / NTILE, BATCH), blk1(THREADS1);
    hipLaunchKernelGGL(ssm_featurize_scan_kernel, g1, blk1, smem1, stream,
                       x, Wdt, bdt, WB, bB, WC, bC, WV, bV, Alog, ybuf);

    // Kernel 2: output projection
    dim3 g2((BATCH * SEQ) / MT2, DIMD / NT2), blk2(THREADS2);
    hipLaunchKernelGGL(out_proj_kernel, g2, blk2, 0, stream, ybuf, Wout, bout, out);
}
